// DeformableFRM_609885356648
// MI455X (gfx1250) — matrix operand order repaired and checked by reading, not compiled, not run on hardware
//
#include <hip/hip_runtime.h>
#include <hip/hip_bf16.h>
#include <math.h>

typedef __bf16 bf16_t;
typedef __attribute__((ext_vector_type(16))) __bf16 v16bf;
typedef __attribute__((ext_vector_type(8)))  __bf16 v8bf;
typedef __attribute__((ext_vector_type(8)))  float  v8f;

#define DIMC   128
#define HH     64
#define WW     64
#define HWSZ   4096
#define BATCH  8
#define D2     256
#define KK2    9

__device__ __forceinline__ bf16_t f2bf(float f) {
    union { float f; unsigned u; } in; in.f = f;
    unsigned u = in.u;
    unsigned r = (u + 0x7FFFu + ((u >> 16) & 1u)) >> 16;   // round-to-nearest-even
    union { unsigned short s; bf16_t b; } out; out.s = (unsigned short)r;
    return out.b;
}

__device__ __forceinline__ v16bf frag_join(const bf16_t* lo, const bf16_t* hi) {
    v8bf a = *(const v8bf*)lo;
    v8bf b = *(const v8bf*)hi;
    v16bf r;
#pragma unroll
    for (int i = 0; i < 8; ++i) { r[i] = a[i]; r[i + 8] = b[i]; }
    return r;
}

// ---------------------------------------------------------------- K0: weights -> bf16
__global__ void k_prep_weights(const float* __restrict__ dc_w,
                               const float* __restrict__ off_w,
                               const float* __restrict__ mk_w,
                               bf16_t* __restrict__ dcw_bf,
                               bf16_t* __restrict__ offmk_bf) {
    int i = blockIdx.x * blockDim.x + threadIdx.x;
    const int N_OFFMK = 32 * 2304;   // rows 0..17 off_w, 18..26 mk_w, 27..31 zero pad
    const int N_DCW   = 128 * 1152;
    if (i < N_OFFMK) {
        float v = 0.f;
        if (i < 18 * 2304)      v = off_w[i];
        else if (i < 27 * 2304) v = mk_w[i - 18 * 2304];
        offmk_bf[i] = f2bf(v);
    }
    int j = i - N_OFFMK;
    if (j >= 0 && j < N_DCW) dcw_bf[j] = f2bf(dc_w[j]);   // (o,i,ky,kx) flat == [o][i*9+k]
}

// ---------------------------------------------------------------- K1: global avg/max pool
__global__ void k_pool(const float* __restrict__ x1, const float* __restrict__ x2,
                       float* __restrict__ y) {
    int bc = blockIdx.x;                 // 0..1023 = (b,c)
    int b = bc >> 7, c = bc & 127;
    const float* p1 = x1 + (size_t)bc * HWSZ;
    const float* p2 = x2 + (size_t)bc * HWSZ;
    float s1 = 0.f, s2 = 0.f, m1 = -3.4e38f, m2 = -3.4e38f;
    for (int i = threadIdx.x; i < HWSZ; i += 256) {
        float a = p1[i], bv = p2[i];
        s1 += a; s2 += bv;
        m1 = fmaxf(m1, a); m2 = fmaxf(m2, bv);
    }
    __shared__ float rs1[256], rs2[256], rm1[256], rm2[256];
    int t = threadIdx.x;
    rs1[t] = s1; rs2[t] = s2; rm1[t] = m1; rm2[t] = m2;
    __syncthreads();
    for (int s = 128; s > 0; s >>= 1) {
        if (t < s) {
            rs1[t] += rs1[t + s]; rs2[t] += rs2[t + s];
            rm1[t] = fmaxf(rm1[t], rm1[t + s]); rm2[t] = fmaxf(rm2[t], rm2[t + s]);
        }
        __syncthreads();
    }
    if (t == 0) {
        y[b * D2 + c]       = rm1[0] + rm2[0];              // max half
        y[b * D2 + 128 + c] = (rs1[0] + rs2[0]) * (1.f / 4096.f); // avg half
    }
}

// ---------------------------------------------------------------- K2: tiny MLP (single block)
__global__ void k_mlp(const float* __restrict__ y,
                      const float* __restrict__ w1, const float* __restrict__ b1,
                      const float* __restrict__ w2, const float* __restrict__ b2,
                      float* __restrict__ cw) {
    __shared__ float sy[BATCH * D2];
    __shared__ float sh[BATCH * D2];
    int t = threadIdx.x;
    for (int i = t; i < BATCH * D2; i += 256) sy[i] = y[i];
    __syncthreads();
    const float* wr1 = w1 + (size_t)t * D2;
    for (int b = 0; b < BATCH; ++b) {
        float acc = b1[t];
        for (int k = 0; k < D2; ++k) acc += sy[b * D2 + k] * wr1[k];
        sh[b * D2 + t] = fmaxf(acc, 0.f);
    }
    __syncthreads();
    const float* wr2 = w2 + (size_t)t * D2;
    for (int b = 0; b < BATCH; ++b) {
        float acc = b2[t];
        for (int k = 0; k < D2; ++k) acc += sh[b * D2 + k] * wr2[k];
        cw[b * D2 + t] = 1.f / (1.f + __expf(-acc));
    }
}

// ---------------------------------------------------------------- K3: recalibration + bf16 concat
__global__ void k_recalib(const float* __restrict__ x1, const float* __restrict__ x2,
                          const float* __restrict__ cw,
                          float* __restrict__ x1c, bf16_t* __restrict__ cat_bf) {
    size_t i = (size_t)blockIdx.x * blockDim.x + threadIdx.x;
    if (i >= (size_t)BATCH * DIMC * HWSZ) return;
    int p = (int)(i & (HWSZ - 1));
    int c = (int)((i >> 12) & 127);
    int b = (int)(i >> 19);
    float s1 = 1.f + 0.5f * cw[b * D2 + c];
    float s2 = 1.f + 0.5f * cw[b * D2 + 128 + c];
    float a  = x1[i] * s1;
    float bb = x2[i] * s2;
    x1c[i] = a;
    size_t base = (size_t)b * D2 * HWSZ;
    cat_bf[base + (size_t)c * HWSZ + p]         = f2bf(a);
    cat_bf[base + (size_t)(128 + c) * HWSZ + p] = f2bf(bb);
}

// ---------------------------------------------------------------- K4: offset/mask conv (implicit GEMM, WMMA)
__global__ void __launch_bounds__(256) k_offmask_conv(
        const bf16_t* __restrict__ cat_bf, const bf16_t* __restrict__ offmk_w,
        const float* __restrict__ off_b, const float* __restrict__ mk_b,
        float* __restrict__ off_out, float* __restrict__ msk_out) {
    __shared__ __attribute__((aligned(16))) bf16_t Csm[WW][296];  // [pixel][K-chunk], padded rows
    int b = blockIdx.x >> 6;
    int h = blockIdx.x & 63;            // pixel tile = one image row
    int tid = threadIdx.x;
    int lane = tid & 31, wave = tid >> 5;
    int mt = wave >> 2, nt = wave & 3;  // 2 M-tiles x 4 N-tiles
    int ln = lane & 15, kh = lane >> 4;
    v8f acc = {};
    const bf16_t* Arow = offmk_w + (size_t)(16 * mt + ln) * 2304;
    for (int g = 0; g < 8; ++g) {       // 8 chunks of 32 channels
        __syncthreads();
        for (int idx = tid; idx < 32 * KK2 * WW; idx += 256) {
            int p  = idx & 63;
            int r  = idx >> 6;          // 0..287 : ci*9 + tap
            int ci = r / 9;
            int kp = r - ci * 9;
            int ky = kp / 3 - 1, kx = kp - (kp / 3) * 3 - 1;
            int hh = h + ky, wp = p + kx;
            bf16_t v = f2bf(0.f);
            if (hh >= 0 && hh < HH && wp >= 0 && wp < WW)
                v = cat_bf[(((size_t)b * D2 + g * 32 + ci) * HH + hh) * WW + wp];
            Csm[p][r] = v;
        }
        __syncthreads();
#pragma unroll
        for (int kk = 0; kk < 288; kk += 32) {
            int gK = g * 288 + kk;
            v16bf afrag = frag_join(Arow + gK + 16 * kh, Arow + gK + 16 * kh + 8);
            const bf16_t* br = &Csm[nt * 16 + ln][kk + 16 * kh];
            v16bf bfrag = frag_join(br, br + 8);
            acc = __builtin_amdgcn_wmma_f32_16x16x32_bf16(false, afrag, false, bfrag,
                                                          (short)0, acc, false, false);
        }
    }
    int col = nt * 16 + ln;
#pragma unroll
    for (int r = 0; r < 8; ++r) {
        int m = 16 * mt + 8 * kh + r;
        float v = acc[r];
        if (m < 18) {
            off_out[(((size_t)b * 18 + m) * HH + h) * WW + col] = v + off_b[m];
        } else if (m < 27) {
            int mm = m - 18;
            msk_out[(((size_t)b * 9 + mm) * HH + h) * WW + col] =
                1.f / (1.f + __expf(-(v + mk_b[mm])));
        }
    }
}

// ---------------------------------------------------------------- K5: fused deformable sampling + GEMM + residual
__global__ void __launch_bounds__(256) k_deform_gemm(
        const float* __restrict__ x1c, const float* __restrict__ x2,
        const float* __restrict__ cw,
        const float* __restrict__ off_out, const float* __restrict__ msk_out,
        const bf16_t* __restrict__ dcw_bf, const float* __restrict__ dc_b,
        float* __restrict__ out) {
    __shared__ int   sIdx[KK2 * WW * 4];                          // 4 gather idx / (tap,pixel)
    __shared__ float sW[KK2 * WW * 4];                            // mask*bilinear weights
    __shared__ __attribute__((aligned(16))) bf16_t Bsm[WW][296];  // sampled B-tile [pixel][K]
    int b = blockIdx.x >> 6;
    int h = blockIdx.x & 63;
    int tid = threadIdx.x;
    int lane = tid & 31, wave = tid >> 5;                          // 8 waves = 8 M-tiles
    int ln = lane & 15, kh = lane >> 4;

    // phase 0: sampling table for this (b,h) row
    for (int e = tid; e < KK2 * WW; e += 256) {
        int k = e >> 6;
        int p = e & 63;
        size_t obase = (((size_t)b * 18 + 2 * k) * HH + h) * WW + p;
        float dy = off_out[obase];
        float dx = off_out[obase + (size_t)HH * WW];
        float mval = msk_out[(((size_t)b * 9 + k) * HH + h) * WW + p];
        float ys = (float)h + (float)(k / 3 - 1) + dy;
        float xs = (float)p + (float)(k % 3 - 1) + dx;
        float y0f = floorf(ys), x0f = floorf(xs);
        float fy = ys - y0f, fx = xs - x0f;
        int y0 = (int)y0f, x0 = (int)x0f;
        float wy[2] = {1.f - fy, fy};
        float wx[2] = {1.f - fx, fx};
#pragma unroll
        for (int jy = 0; jy < 2; ++jy)
#pragma unroll
            for (int jx = 0; jx < 2; ++jx) {
                int yy = y0 + jy, xx = x0 + jx;
                bool valid = (yy >= 0) && (yy < HH) && (xx >= 0) && (xx < WW);
                int yc = min(max(yy, 0), HH - 1), xc = min(max(xx, 0), WW - 1);
                int j = jy * 2 + jx;
                sIdx[e * 4 + j] = yc * WW + xc;
                sW[e * 4 + j]   = valid ? wy[jy] * wx[jx] * mval : 0.f;
            }
    }

    v8f acc[4] = {};                               // 4 N-tiles per wave
    const bf16_t* Arow = dcw_bf + (size_t)(16 * wave + ln) * 1152;
    for (int g = 0; g < 4; ++g) {                  // 4 chunks of 32 channels (K-chunk = 288)
        __syncthreads();
        for (int idx = tid; idx < 32 * KK2 * WW; idx += 256) {
            int p  = idx & 63;
            int r  = idx >> 6;                     // ci*9 + tap
            int ci = r / 9;
            int kp = r - ci * 9;
            const float* plane = x1c + ((size_t)b * DIMC + g * 32 + ci) * HWSZ;
            int e4 = (kp * WW + p) * 4;
            float v = sW[e4 + 0] * plane[sIdx[e4 + 0]]
                    + sW[e4 + 1] * plane[sIdx[e4 + 1]]
                    + sW[e4 + 2] * plane[sIdx[e4 + 2]]
                    + sW[e4 + 3] * plane[sIdx[e4 + 3]];
            Bsm[p][r] = f2bf(v);
        }
        __syncthreads();
#pragma unroll
        for (int kk = 0; kk < 288; kk += 32) {
            int gK = g * 288 + kk;
            v16bf afrag = frag_join(Arow + gK + 16 * kh, Arow + gK + 16 * kh + 8);
#pragma unroll
            for (int nt = 0; nt < 4; ++nt) {
                const bf16_t* br = &Bsm[nt * 16 + ln][kk + 16 * kh];
                v16bf bfrag = frag_join(br, br + 8);
                acc[nt] = __builtin_amdgcn_wmma_f32_16x16x32_bf16(false, afrag, false, bfrag,
                                                                  (short)0, acc[nt], false, false);
            }
        }
    }

    // epilogue: + dc_b + x2_c residual (recomputed from x2 and cw)
#pragma unroll
    for (int nt = 0; nt < 4; ++nt) {
        int col = nt * 16 + ln;
#pragma unroll
        for (int r = 0; r < 8; ++r) {
            int o = 16 * wave + 8 * kh + r;
            float s2 = 1.f + 0.5f * cw[b * D2 + 128 + o];
            size_t oidx = (((size_t)b * DIMC + o) * HH + h) * WW + col;
            out[oidx] = acc[nt][r] + dc_b[o] + x2[oidx] * s2;
        }
    }
}

// ---------------------------------------------------------------- host launcher
extern "C" void kernel_launch(void* const* d_in, const int* in_sizes, int n_in,
                              void* d_out, int out_size, void* d_ws, size_t ws_size,
                              hipStream_t stream) {
    (void)in_sizes; (void)n_in; (void)out_size; (void)ws_size;
    const float* x1    = (const float*)d_in[0];
    const float* x2    = (const float*)d_in[1];
    const float* w1    = (const float*)d_in[2];
    const float* b1    = (const float*)d_in[3];
    const float* w2    = (const float*)d_in[4];
    const float* b2    = (const float*)d_in[5];
    const float* dc_w  = (const float*)d_in[6];
    const float* dc_b  = (const float*)d_in[7];
    const float* off_w = (const float*)d_in[8];
    const float* off_b = (const float*)d_in[9];
    const float* mk_w  = (const float*)d_in[10];
    const float* mk_b  = (const float*)d_in[11];
    float* out = (float*)d_out;

    char* ws = (char*)d_ws;
    size_t off = 0;
    auto alloc = [&](size_t bytes) -> void* {
        void* p = (void*)(ws + off);
        off += (bytes + 255) & ~(size_t)255;
        return p;
    };
    float*  y        = (float*) alloc((size_t)BATCH * D2 * 4);
    float*  cwbuf    = (float*) alloc((size_t)BATCH * D2 * 4);
    bf16_t* dcw_bf   = (bf16_t*)alloc((size_t)128 * 1152 * 2);
    bf16_t* offmk_bf = (bf16_t*)alloc((size_t)32 * 2304 * 2);
    float*  x1c      = (float*) alloc((size_t)BATCH * DIMC * HWSZ * 4);
    bf16_t* cat_bf   = (bf16_t*)alloc((size_t)BATCH * D2 * HWSZ * 2);
    float*  off_out  = (float*) alloc((size_t)BATCH * 18 * HWSZ * 4);
    float*  msk_out  = (float*) alloc((size_t)BATCH * 9 * HWSZ * 4);

    k_prep_weights<<<(32 * 2304 + 128 * 1152 + 255) / 256, 256, 0, stream>>>(
        dc_w, off_w, mk_w, dcw_bf, offmk_bf);
    k_pool<<<BATCH * DIMC, 256, 0, stream>>>(x1, x2, y);
    k_mlp<<<1, 256, 0, stream>>>(y, w1, b1, w2, b2, cwbuf);
    k_recalib<<<(BATCH * DIMC * HWSZ + 255) / 256, 256, 0, stream>>>(x1, x2, cwbuf, x1c, cat_bf);
    k_offmask_conv<<<BATCH * HH, 256, 0, stream>>>(cat_bf, offmk_bf, off_b, mk_b, off_out, msk_out);
    k_deform_gemm<<<BATCH * HH, 256, 0, stream>>>(x1c, x2, cwbuf, off_out, msk_out, dcw_bf, dc_b, out);
}
